// Decoder_48309792145988
// MI455X (gfx1250) — matrix operand order repaired and checked by reading, compiled, not run on hardware
//
#include <hip/hip_runtime.h>
#include <hip/hip_bf16.h>

// ---------------------------------------------------------------------------
// Tacotron-style decoder for MI455X (gfx1250, wave32, WMMA).
//
// Strategy:
//  * One-time prep kernels: convert all weight matrices (and encoder outputs)
//    to f16 in workspace; compute processed_memory = enc @ mem_w.T with a
//    grid-wide WMMA GEMM (6144x256x256).
//  * One persistent 1024-thread (32-wave) workgroup runs the 200-step scan.
//    All recurrent state and activation staging lives in LDS (~230 KB,
//    dynamic). Every GEMM in the step is M=16, done with
//    v_wmma_f32_16x16x32_f16, one wave per 16x16 output tile.
//  * Elementwise gates / Bahdanau energy use native TRANS ops
//    (v_tanh_f32 when available, v_exp_f32 + v_rcp_f32 otherwise).
// ---------------------------------------------------------------------------

typedef __attribute__((ext_vector_type(16))) _Float16 v16h;
typedef __attribute__((ext_vector_type(8)))  _Float16 v8h;
typedef __attribute__((ext_vector_type(8)))  float    v8f;

#define NT 1024  // decoder block threads (32 waves of 32)

// ---------------------------- math helpers ---------------------------------
__device__ __forceinline__ float frcp(float x) {
    return __builtin_amdgcn_rcpf(x);                // v_rcp_f32
}
__device__ __forceinline__ float fsigmoid(float x) {
    return frcp(1.0f + __expf(-x));                 // v_exp_f32 + v_rcp_f32
}
__device__ __forceinline__ float ftanh(float x) {
#if __has_builtin(__builtin_amdgcn_tanhf)
    return __builtin_amdgcn_tanhf(x);               // native v_tanh_f32 (TRANS)
#else
    float e = __expf(-2.0f * fabsf(x));
    float r = (1.0f - e) * frcp(1.0f + e);
    return x >= 0.0f ? r : -r;
#endif
}

// ---------------------------- WMMA helpers ---------------------------------
// A-fragment (16x32 f16, ISA 7.12.2): lane m = lane&15, half-group g = lane>>4
//   a[0..7]  = A[m][k0 + g*8 .. +7]
//   a[8..15] = A[m][k0 + 16 + g*8 .. +7]
__device__ __forceinline__ v16h load_a(const _Float16* A, int lda, int k0, int lane) {
    int m = lane & 15, g = (lane >> 4) & 1;
    const _Float16* p = A + (size_t)m * lda + k0 + g * 8;
    v8h lo = *(const v8h*)p;
    v8h hi = *(const v8h*)(p + 16);
    return __builtin_shufflevector(lo, hi, 0, 1, 2, 3, 4, 5, 6, 7,
                                           8, 9, 10, 11, 12, 13, 14, 15);
}
// B-fragment (32x16 f16): B[k][n] = W[n0+n][k] from row-major W(out,in).
//   lane n = lane&15, group g = lane>>4: b[0..7] = K k0+g*8..+7, b[8..15] = K k0+16+g*8..+7
__device__ __forceinline__ v16h load_b(const _Float16* W, int ldw, int n0, int k0, int lane) {
    int n = lane & 15, g = (lane >> 4) & 1;
    const _Float16* p = W + (size_t)(n0 + n) * ldw + k0 + g * 8;
    v8h lo = *(const v8h*)p;
    v8h hi = *(const v8h*)(p + 16);
    return __builtin_shufflevector(lo, hi, 0, 1, 2, 3, 4, 5, 6, 7,
                                           8, 9, 10, 11, 12, 13, 14, 15);
}
__device__ __forceinline__ v8f wmma16(v16h a, v16h b, v8f c) {
    return __builtin_amdgcn_wmma_f32_16x16x32_f16(false, a, false, b,
                                                  (short)0, c, false, false);
}
// One 16x16 output tile: D += A(16xK) * W.T(Kx16) at columns [n0, n0+16)
__device__ __forceinline__ void gemm_rowtile(const _Float16* A, int lda,
                                             const _Float16* W, int ldw,
                                             int n0, int K, int lane, v8f& acc) {
    for (int k0 = 0; k0 < K; k0 += 32) {
        v16h a = load_a(A, lda, k0, lane);
        v16h b = load_b(W, ldw, n0, k0, lane);
        acc = wmma16(a, b, acc);
    }
}

// ---------------------------- GRU cell (B=16, H=256) ------------------------
// Processes gates r,z,n sequentially; each gate: 16 N-tiles of gi (waves 0-15)
// and 16 N-tiles of gh (waves 16-31) -> exactly one tile per wave.
__device__ void gru_cell(const _Float16* xin16, int Kin,
                         const _Float16* hin16,
                         const _Float16* Wih, const _Float16* Whh,
                         const float* bih, const float* bhh,
                         float* h, _Float16* h16,
                         float* S1, float* S2, float* rbuf, float* zbuf,
                         int tid, int lane, int wave) {
    for (int gate = 0; gate < 3; ++gate) {
        if (wave < 16) {
            int n0 = wave * 16;
            v8f acc = {};
            gemm_rowtile(xin16, Kin, Wih + (size_t)(gate * 256) * Kin, Kin, n0, Kin, lane, acc);
            int n = lane & 15, g = (lane >> 4) & 1;
            float bias = bih[gate * 256 + n0 + n];
#pragma unroll
            for (int r = 0; r < 8; ++r) S1[(g * 8 + r) * 256 + n0 + n] = acc[r] + bias;
        } else {
            int n0 = (wave - 16) * 16;
            v8f acc = {};
            gemm_rowtile(hin16, 256, Whh + (size_t)(gate * 256) * 256, 256, n0, 256, lane, acc);
            int n = lane & 15, g = (lane >> 4) & 1;
            float bias = bhh[gate * 256 + n0 + n];
#pragma unroll
            for (int r = 0; r < 8; ++r) S2[(g * 8 + r) * 256 + n0 + n] = acc[r] + bias;
        }
        __syncthreads();
        for (int i = tid; i < 4096; i += NT) {
            if (gate == 0)      rbuf[i] = fsigmoid(S1[i] + S2[i]);
            else if (gate == 1) zbuf[i] = fsigmoid(S1[i] + S2[i]);
            else {
                float nn = ftanh(S1[i] + rbuf[i] * S2[i]);
                float z  = zbuf[i];
                float hv = (1.0f - z) * nn + z * h[i];
                h[i]   = hv;
                h16[i] = (_Float16)hv;
            }
        }
        __syncthreads();
    }
}

// ---------------------------- prep kernels ----------------------------------
__global__ void cvt_f16_kernel(_Float16* __restrict__ dst, const float* __restrict__ src,
                               int rows, int scols, int dcols) {
    int i = blockIdx.x * blockDim.x + threadIdx.x;
    if (i >= rows * dcols) return;
    int r = i / dcols, c = i - r * dcols;
    dst[i] = (_Float16)((c < scols) ? src[(size_t)r * scols + c] : 0.0f);
}

// processed_memory = enc(6144x256) @ mem_w.T(256x256). 6144 tiles, 1 per wave.
__global__ __launch_bounds__(256) void pm_gemm_kernel(const _Float16* __restrict__ ench,
                                                      const _Float16* __restrict__ memwh,
                                                      float* __restrict__ P) {
    int lane = threadIdx.x & 31;
    int wave = threadIdx.x >> 5;
    int tile = blockIdx.x * 8 + wave;     // 384 row-tiles * 16 col-tiles
    int mt = tile >> 4, nt = tile & 15;
    const _Float16* A = ench + (size_t)mt * 16 * 256;
    v8f acc = {};
    gemm_rowtile(A, 256, memwh, 256, nt * 16, 256, lane, acc);
    int n = lane & 15, g = (lane >> 4) & 1;
#pragma unroll
    for (int r = 0; r < 8; ++r)
        P[((size_t)mt * 16 + g * 8 + r) * 256 + nt * 16 + n] = acc[r];
}

// ---------------------------- persistent decoder -----------------------------
__global__ void __launch_bounds__(NT, 1)
tts_decoder_kernel(const float* __restrict__ enc,      // (16,384,256)
                   const float* __restrict__ x_in,     // (16,200,400)
                   const int*   __restrict__ memlen,   // (16)
                   const float* __restrict__ p_b1,     // 256
                   const float* __restrict__ p_b2,     // 128
                   const float* __restrict__ ag_bih,   // 768
                   const float* __restrict__ ag_bhh,   // 768
                   const float* __restrict__ attn_v,   // 256
                   const float* __restrict__ proj_b,   // 256
                   const float* __restrict__ d_bih,    // (2,768)
                   const float* __restrict__ d_bhh,    // (2,768)
                   const float* __restrict__ mel_b,    // 400
                   const _Float16* __restrict__ W1h,   // 256x416 (K padded)
                   const _Float16* __restrict__ W2h,   // 128x256
                   const _Float16* __restrict__ AGIh,  // 768x384
                   const _Float16* __restrict__ AGHh,  // 768x256
                   const _Float16* __restrict__ Qh,    // 256x256
                   const _Float16* __restrict__ PIh,   // 256x512
                   const _Float16* __restrict__ DIh,   // 2x768x256
                   const _Float16* __restrict__ DHh,   // 2x768x256
                   const _Float16* __restrict__ MELh,  // 400x256
                   const float* __restrict__ P,        // (16,384,256)
                   float* __restrict__ out_mel,        // (16,200,400)
                   float* __restrict__ out_align) {    // (16,200,384)
    extern __shared__ char smem_raw[];
    float* fbase     = (float*)smem_raw;
    float* attn_h    = fbase;             // 16x256
    float* attention = attn_h + 4096;     // 16x256
    float* d1        = attention + 4096;
    float* d2        = d1 + 4096;
    float* S1        = d2 + 4096;
    float* S2        = S1 + 4096;
    float* rbuf      = S2 + 4096;
    float* zbuf      = rbuf + 4096;
    float* dec_in    = zbuf + 4096;
    float* ebuf      = dec_in + 4096;     // 16x384
    float* vA        = ebuf + 6144;       // 256
    _Float16* A0h = (_Float16*)(vA + 256);   // 16x512 (x staging / cat staging)
    _Float16* A1h = A0h + 16 * 512;          // 16x256 (pn1 / dec_in staging)
    _Float16* A2h = A1h + 16 * 256;          // 16x384 (attn-GRU input staging)
    _Float16* hA  = A2h + 16 * 384;          // f16 mirror of attn_h
    _Float16* d1h = hA + 4096;
    _Float16* d2h = d1h + 4096;
    int* mlen = (int*)(d2h + 4096);

    const int tid = threadIdx.x, lane = tid & 31, wave = tid >> 5;

    // init state
    for (int i = tid; i < 4096; i += NT) {
        attn_h[i] = 0.0f; attention[i] = 0.0f; d1[i] = 0.0f; d2[i] = 0.0f;
        hA[i] = (_Float16)0.0f; d1h[i] = (_Float16)0.0f; d2h[i] = (_Float16)0.0f;
    }
    for (int i = tid; i < 256; i += NT) vA[i] = attn_v[i];
    if (tid < 16) mlen[tid] = memlen[tid];
    __syncthreads();

    for (int t = 0; t < 200; ++t) {
        // P0: stage teacher-forced frame (K padded 400->416)
        for (int i = tid; i < 16 * 416; i += NT) {
            int b = i / 416, k = i - b * 416;
            float v = 0.0f;
            if (k < 400 && t > 0) v = x_in[((size_t)b * 200 + (t - 1)) * 400 + k];
            A0h[b * 416 + k] = (_Float16)v;
        }
        __syncthreads();
        // P1: prenet1 (16x416 * 416x256) -> relu -> A1h
        if (wave < 16) {
            int n0 = wave * 16;
            v8f acc = {};
            gemm_rowtile(A0h, 416, W1h, 416, n0, 416, lane, acc);
            int n = lane & 15, g = (lane >> 4) & 1;
            float bias = p_b1[n0 + n];
#pragma unroll
            for (int r = 0; r < 8; ++r) {
                float v = fmaxf(acc[r] + bias, 0.0f);
                A1h[(g * 8 + r) * 256 + n0 + n] = (_Float16)v;
            }
        }
        __syncthreads();
        // P2: prenet2 -> A2h[:,0:128]; copy prev attention -> A2h[:,128:384]
        if (wave < 8) {
            int n0 = wave * 16;
            v8f acc = {};
            gemm_rowtile(A1h, 256, W2h, 256, n0, 256, lane, acc);
            int n = lane & 15, g = (lane >> 4) & 1;
            float bias = p_b2[n0 + n];
#pragma unroll
            for (int r = 0; r < 8; ++r) {
                float v = fmaxf(acc[r] + bias, 0.0f);
                A2h[(g * 8 + r) * 384 + n0 + n] = (_Float16)v;
            }
        } else if (tid >= 512) {
            for (int i = tid - 512; i < 4096; i += 512) {
                int b = i >> 8, j = i & 255;
                A2h[b * 384 + 128 + j] = (_Float16)attention[i];
            }
        }
        __syncthreads();
        // P3: attention GRU cell
        gru_cell(A2h, 384, hA, AGIh, AGHh, ag_bih, ag_bhh,
                 attn_h, hA, S1, S2, rbuf, zbuf, tid, lane, wave);
        // P4: pq = attn_h @ q_w.T -> S1
        if (wave < 16) {
            int n0 = wave * 16;
            v8f acc = {};
            gemm_rowtile(hA, 256, Qh, 256, n0, 256, lane, acc);
            int n = lane & 15, g = (lane >> 4) & 1;
#pragma unroll
            for (int r = 0; r < 8; ++r) S1[(g * 8 + r) * 256 + n0 + n] = acc[r];
        }
        __syncthreads();
        // P5: e[b,tt] = sum_k tanh(pq[b,k] + P[b,tt,k]) * v[k], with pad mask
        for (int row = tid; row < 16 * 384; row += NT) {
            int b = row / 384, tt = row - b * 384;
            const float* Pp = P + ((size_t)b * 384 + tt) * 256;
            const float* pq = S1 + b * 256;
            float acc = 0.0f;
#pragma unroll 8
            for (int k = 0; k < 256; ++k) acc += ftanh(pq[k] + Pp[k]) * vA[k];
            ebuf[row] = (tt < mlen[b]) ? acc : -1e30f;
        }
        __syncthreads();
        // P6: softmax over T_enc, one wave per batch row (384 = 12*32)
        if (wave < 16) {
            int b = wave;
            float m = -1e30f;
#pragma unroll
            for (int i = 0; i < 12; ++i) m = fmaxf(m, ebuf[b * 384 + lane + 32 * i]);
#pragma unroll
            for (int off = 16; off > 0; off >>= 1) m = fmaxf(m, __shfl_xor(m, off, 32));
            float s = 0.0f, vals[12];
#pragma unroll
            for (int i = 0; i < 12; ++i) {
                float ex = __expf(ebuf[b * 384 + lane + 32 * i] - m);
                vals[i] = ex; s += ex;
            }
#pragma unroll
            for (int off = 16; off > 0; off >>= 1) s += __shfl_xor(s, off, 32);
            float inv = frcp(s);
#pragma unroll
            for (int i = 0; i < 12; ++i) ebuf[b * 384 + lane + 32 * i] = vals[i] * inv;
        }
        __syncthreads();
        // P7: context = align @ enc ; build cat(attn_h, attention) staging
        for (int i = tid; i < 4096; i += NT) {
            int b = i >> 8, d = i & 255;
            const float* al = ebuf + b * 384;
            const float* ep = enc + (size_t)b * 384 * 256 + d;
            float acc = 0.0f;
#pragma unroll 4
            for (int tt = 0; tt < 384; ++tt) acc += al[tt] * ep[(size_t)tt * 256];
            attention[i] = acc;
            A0h[b * 512 + 256 + d] = (_Float16)acc;
            A0h[b * 512 + d]       = hA[i];
        }
        __syncthreads();
        // P8: dec_in = cat @ proj_in_w.T + b
        if (wave < 16) {
            int n0 = wave * 16;
            v8f acc = {};
            gemm_rowtile(A0h, 512, PIh, 512, n0, 512, lane, acc);
            int n = lane & 15, g = (lane >> 4) & 1;
            float bias = proj_b[n0 + n];
#pragma unroll
            for (int r = 0; r < 8; ++r) {
                float v = acc[r] + bias;
                dec_in[(g * 8 + r) * 256 + n0 + n] = v;
                A1h[(g * 8 + r) * 256 + n0 + n]    = (_Float16)v;
            }
        }
        __syncthreads();
        // P9: two residual decoder GRU cells
        for (int c = 0; c < 2; ++c) {
            float*     hc   = (c == 0) ? d1 : d2;
            _Float16*  hc16 = (c == 0) ? d1h : d2h;
            gru_cell(A1h, 256, hc16,
                     DIh + (size_t)c * 768 * 256, DHh + (size_t)c * 768 * 256,
                     d_bih + c * 768, d_bhh + c * 768,
                     hc, hc16, S1, S2, rbuf, zbuf, tid, lane, wave);
            for (int i = tid; i < 4096; i += NT) {
                float v = hc[i] + dec_in[i];
                dec_in[i] = v;
                A1h[i]    = (_Float16)v;
            }
            __syncthreads();
        }
        // P10: mel projection (25 N-tiles) + alignment write-out
        if (wave < 25) {
            int n0 = wave * 16;
            v8f acc = {};
            gemm_rowtile(A1h, 256, MELh, 256, n0, 256, lane, acc);
            int n = lane & 15, g = (lane >> 4) & 1;
            float bias = mel_b[n0 + n];
#pragma unroll
            for (int r = 0; r < 8; ++r) {
                int b = g * 8 + r;
                out_mel[((size_t)b * 200 + t) * 400 + n0 + n] = acc[r] + bias;
            }
        } else {
            for (int i = tid - 800; i < 6144; i += 224) {
                int b = i / 384, tt = i - b * 384;
                out_align[((size_t)b * 200 + t) * 384 + tt] = ebuf[i];
            }
        }
        __syncthreads();
    }
}

// ---------------------------- host launch -----------------------------------
extern "C" void kernel_launch(void* const* d_in, const int* in_sizes, int n_in,
                              void* d_out, int out_size, void* d_ws, size_t ws_size,
                              hipStream_t stream) {
    (void)in_sizes; (void)n_in; (void)out_size; (void)ws_size;

    const float* enc    = (const float*)d_in[0];
    const float* xin    = (const float*)d_in[1];
    const int*   memlen = (const int*)d_in[2];
    const float* p_w1   = (const float*)d_in[3];
    const float* p_b1   = (const float*)d_in[4];
    const float* p_w2   = (const float*)d_in[5];
    const float* p_b2   = (const float*)d_in[6];
    const float* mem_w  = (const float*)d_in[7];
    const float* ag_wih = (const float*)d_in[8];
    const float* ag_whh = (const float*)d_in[9];
    const float* ag_bih = (const float*)d_in[10];
    const float* ag_bhh = (const float*)d_in[11];
    const float* q_w    = (const float*)d_in[12];
    const float* a_v    = (const float*)d_in[13];
    const float* pi_w   = (const float*)d_in[14];
    const float* pi_b   = (const float*)d_in[15];
    const float* dwih   = (const float*)d_in[16];
    const float* dwhh   = (const float*)d_in[17];
    const float* dbih   = (const float*)d_in[18];
    const float* dbhh   = (const float*)d_in[19];
    const float* mel_w  = (const float*)d_in[20];
    const float* mel_b  = (const float*)d_in[21];

    float* out_mel   = (float*)d_out;                 // (16,200,400)
    float* out_align = out_mel + (size_t)16 * 200 * 400;  // (16,200,384)

    // workspace carve (~13 MB)
    char*  ws  = (char*)d_ws;
    size_t off = 0;
    auto carve = [&](size_t bytes) -> char* {
        char* p = ws + off;
        off = (off + bytes + 255) & ~(size_t)255;
        return p;
    };
    _Float16* ench  = (_Float16*)carve((size_t)16 * 384 * 256 * 2);
    float*    P     = (float*)   carve((size_t)16 * 384 * 256 * 4);
    _Float16* W1h   = (_Float16*)carve((size_t)256 * 416 * 2);
    _Float16* W2h   = (_Float16*)carve((size_t)128 * 256 * 2);
    _Float16* AGIh  = (_Float16*)carve((size_t)768 * 384 * 2);
    _Float16* AGHh  = (_Float16*)carve((size_t)768 * 256 * 2);
    _Float16* Qh    = (_Float16*)carve((size_t)256 * 256 * 2);
    _Float16* PIh   = (_Float16*)carve((size_t)256 * 512 * 2);
    _Float16* DIh   = (_Float16*)carve((size_t)2 * 768 * 256 * 2);
    _Float16* DHh   = (_Float16*)carve((size_t)2 * 768 * 256 * 2);
    _Float16* MELh  = (_Float16*)carve((size_t)400 * 256 * 2);
    _Float16* MEMWh = (_Float16*)carve((size_t)256 * 256 * 2);

    auto cvt = [&](_Float16* dst, const float* src, int rows, int scols, int dcols) {
        int total = rows * dcols;
        cvt_f16_kernel<<<(total + 255) / 256, 256, 0, stream>>>(dst, src, rows, scols, dcols);
    };
    cvt(ench,  enc,    16 * 384, 256, 256);
    cvt(W1h,   p_w1,   256, 400, 416);   // pad K 400 -> 416 with zeros
    cvt(W2h,   p_w2,   128, 256, 256);
    cvt(AGIh,  ag_wih, 768, 384, 384);
    cvt(AGHh,  ag_whh, 768, 256, 256);
    cvt(Qh,    q_w,    256, 256, 256);
    cvt(PIh,   pi_w,   256, 512, 512);
    cvt(DIh,   dwih,   2 * 768, 256, 256);
    cvt(DHh,   dwhh,   2 * 768, 256, 256);
    cvt(MELh,  mel_w,  400, 256, 256);
    cvt(MEMWh, mem_w,  256, 256, 256);

    // processed_memory GEMM: 6144 tiles, 8 waves/block -> 768 blocks
    pm_gemm_kernel<<<768, 256, 0, stream>>>(ench, MEMWh, P);

    // persistent decoder: 1 block, 32 waves, ~230 KB dynamic LDS
    const size_t SMEM_BYTES =
        (size_t)(9 * 4096 + 6144 + 256) * 4 +          // f32 regions
        (size_t)(16 * 512 + 16 * 256 + 16 * 384 + 3 * 4096) * 2 +  // f16 regions
        64;                                            // mlen ints
    static int smem_attr_set = 0;  // host-side, deterministic, capture-safe
    if (!smem_attr_set) {
        hipFuncSetAttribute(reinterpret_cast<const void*>(tts_decoder_kernel),
                            hipFuncAttributeMaxDynamicSharedMemorySize,
                            (int)SMEM_BYTES);
        smem_attr_set = 1;
    }
    tts_decoder_kernel<<<1, NT, SMEM_BYTES, stream>>>(
        enc, xin, memlen, p_b1, p_b2, ag_bih, ag_bhh, a_v, pi_b,
        dbih, dbhh, mel_b,
        W1h, W2h, AGIh, AGHh, Qh, PIh, DIh, DHh, MELh, P,
        out_mel, out_align);
}
